// CrystalHypergraphConv_16020228014459
// MI455X (gfx1250) — matrix operand order repaired and checked by reading, not compiled, not run on hardware
//
#include <hip/hip_runtime.h>
#include <hip/hip_bf16.h>

typedef _Float16 half_t;
typedef __attribute__((ext_vector_type(16))) _Float16 v16h;
typedef __attribute__((ext_vector_type(8)))  float    v8f;

union Frag16 { v16h h; uint4 q[2]; };

#define HD 64
#define HF 35
#define ZD 163
#define XK 92

__device__ __forceinline__ float softplusf(float x) {
    return fmaxf(x, 0.0f) + log1pf(expf(-fabsf(x)));
}
__device__ __forceinline__ float sigmoidf(float x) {
    return 1.0f / (1.0f + expf(-x));
}
__device__ __forceinline__ unsigned int pack2h(float lo, float hi) {
    union { _Float16 h[2]; unsigned int u; } p;
    p.h[0] = (half_t)lo;
    p.h[1] = (half_t)hi;
    return p.u;
}

// ---------------- generic helpers ----------------

__global__ void fzero_kernel(float* __restrict__ p, long long n) {
    long long i = (long long)blockIdx.x * blockDim.x + threadIdx.x;
    if (i < n) p[i] = 0.0f;
}

__global__ void inv_inplace_kernel(float* __restrict__ p, int n) {
    int i = blockIdx.x * blockDim.x + threadIdx.x;
    if (i < n) p[i] = 1.0f / fmaxf(p[i], 1.0f);
}

__global__ void count_inc_kernel(const int* __restrict__ node_idx,
                                 const int* __restrict__ hedge_idx,
                                 float* __restrict__ cnt_node,
                                 float* __restrict__ cnt_hedge, int E) {
    int i = blockIdx.x * blockDim.x + threadIdx.x;
    if (i < E) {
        atomicAdd(&cnt_node[node_idx[i]], 1.0f);
        atomicAdd(&cnt_hedge[hedge_idx[i]], 1.0f);
    }
}

__global__ void count_batch_kernel(const int* __restrict__ batch,
                                   float* __restrict__ cnt, int N) {
    int i = blockIdx.x * blockDim.x + threadIdx.x;
    if (i < N) atomicAdd(&cnt[batch[i]], 1.0f);
}

// ---------------- fragment packing kernels ----------------
// B-fragment-major weight pack: out[kk][nb][lane][v] (dwords), from W[K,64] f32.
// ISA dense 16-bit B layout: V0-3 -> K 0..7 / 8..15 (by lane half), V4-7 -> +16, 2 K per VGPR.
__global__ void cvt_w_pack_kernel(const float* __restrict__ W, unsigned int* __restrict__ out,
                                  int K, int kSteps) {
    int i = blockIdx.x * blockDim.x + threadIdx.x;
    if (i < kSteps * 1024) {
        int v    = i & 7;
        int lane = (i >> 3) & 31;
        int nb   = (i >> 8) & 3;
        int kk   = i >> 10;
        int n    = nb * 16 + (lane & 15);
        int kb   = kk * 32 + (v >> 2) * 16 + (lane >> 4) * 8 + (v & 3) * 2;
        float lo = (kb     < K) ? W[(size_t)kb * HD + n]       : 0.0f;
        float hi = (kb + 1 < K) ? W[(size_t)(kb + 1) * HD + n] : 0.0f;
        out[i] = pack2h(lo, hi);
    }
}

// A-fragment-major pack of x: out[tile][kk(0..2)][lane][v], from x[N,92] f32.
// ISA 16-bit A layout: V0-3 -> K 0..7 / 8..15 (by lane half), V4-7 -> +16.
__global__ void cvt_x_pack_kernel(const float* __restrict__ x, unsigned int* __restrict__ out,
                                  int N, long long total) {
    long long i = (long long)blockIdx.x * blockDim.x + threadIdx.x;
    if (i < total) {
        int v    = (int)(i & 7);
        int lane = (int)(i >> 3) & 31;
        int kk   = (int)((i >> 8) % 3);
        long long tile = i / 768;
        int l16 = lane & 15, h16 = lane >> 4;
        long long row = tile * 16 + l16;
        int kb = kk * 32 + (v >> 2) * 16 + h16 * 8 + (v & 3) * 2;
        float lo = 0.0f, hi = 0.0f;
        if (row < N) {
            if (kb     < XK) lo = x[row * XK + kb];
            if (kb + 1 < XK) hi = x[row * XK + kb + 1];
        }
        out[i] = pack2h(lo, hi);
    }
}

// A-fragment-major pack of z = [h | za/deg | zb/deg | pad] : out[tile][kk(0..5)][lane][v]
__global__ void build_z_pack_kernel(const float* __restrict__ h,
                                    const float* __restrict__ za,
                                    const float* __restrict__ zb,
                                    const float* __restrict__ inv_node,
                                    unsigned int* __restrict__ out,
                                    int N, long long total) {
    long long i = (long long)blockIdx.x * blockDim.x + threadIdx.x;
    if (i < total) {
        int v    = (int)(i & 7);
        int lane = (int)(i >> 3) & 31;
        int kk   = (int)((i >> 8) % 6);
        long long tile = i / 1536;
        int l16 = lane & 15, h16 = lane >> 4;
        long long row = tile * 16 + l16;
        int kb = kk * 32 + (v >> 2) * 16 + h16 * 8 + (v & 3) * 2;
        float lo = 0.0f, hi = 0.0f;
        if (row < N) {
            float inv = inv_node[row];
            #pragma unroll
            for (int t = 0; t < 2; ++t) {
                int c = kb + t;
                float val;
                if (c < HD)           val = h[row * HD + c];
                else if (c < HD + HF) val = za[row * HF + (c - HD)] * inv;
                else if (c < ZD)      val = zb[row * HD + (c - HD - HF)] * inv;
                else                  val = 0.0f;
                if (t == 0) lo = val; else hi = val;
            }
        }
        out[i] = pack2h(lo, hi);
    }
}

// ---------------- scatter / segment kernels ----------------

__global__ void scatter_hm_kernel(const int* __restrict__ node_idx,
                                  const int* __restrict__ hedge_idx,
                                  const float* __restrict__ h,
                                  float* __restrict__ hm, long long total) {
    long long i = (long long)blockIdx.x * blockDim.x + threadIdx.x;
    if (i < total) {
        int j = (int)(i & 63);
        long long e = i >> 6;
        int nd = node_idx[e], hg = hedge_idx[e];
        atomicAdd(&hm[(size_t)hg * HD + j], h[(size_t)nd * HD + j]);
    }
}

__global__ void rowscale_kernel(float* __restrict__ p, const float* __restrict__ inv,
                                long long n, int cols) {
    long long i = (long long)blockIdx.x * blockDim.x + threadIdx.x;
    if (i < n) p[i] *= inv[i / cols];
}

__global__ void scatter_z_kernel(const int* __restrict__ node_idx,
                                 const int* __restrict__ hedge_idx,
                                 const float* __restrict__ hedge_attr,
                                 const float* __restrict__ hm,
                                 float* __restrict__ za, float* __restrict__ zb,
                                 long long total) {
    long long i = (long long)blockIdx.x * blockDim.x + threadIdx.x;
    if (i < total) {
        int j = (int)(i & 127);
        long long e = i >> 7;
        if (j < HF) {
            int nd = node_idx[e], hg = hedge_idx[e];
            atomicAdd(&za[(size_t)nd * HF + j], hedge_attr[(size_t)hg * HF + j]);
        } else if (j < HF + HD) {
            int nd = node_idx[e], hg = hedge_idx[e];
            int jj = j - HF;
            atomicAdd(&zb[(size_t)nd * HD + jj], hm[(size_t)hg * HD + jj]);
        }
    }
}

__global__ void pool_scatter_kernel(const int* __restrict__ batch,
                                    const float* __restrict__ h,
                                    float* __restrict__ pooled, long long total) {
    long long i = (long long)blockIdx.x * blockDim.x + threadIdx.x;
    if (i < total) {
        int j = (int)(i & 63);
        long long r = i >> 6;
        atomicAdd(&pooled[(size_t)batch[r] * HD + j], h[i]);
    }
}

__global__ void g_kernel(const float* __restrict__ pooled,
                         const float* __restrict__ inv_batch,
                         const float* __restrict__ proj_w,
                         const float* __restrict__ proj_b,
                         float* __restrict__ g, int total /* NG*128 */) {
    int i = blockIdx.x * blockDim.x + threadIdx.x;
    if (i < total) {
        int c = i & 127;
        int gi = i >> 7;
        float inv = inv_batch[gi];
        float s = proj_b[c];
        #pragma unroll 8
        for (int k = 0; k < HD; ++k)
            s += pooled[gi * HD + k] * inv * proj_w[k * 128 + c];
        g[i] = softplusf(s);
    }
}

__global__ void out_kernel(const float* __restrict__ g,
                           const float* __restrict__ out_w,
                           const float* __restrict__ out_b,
                           float* __restrict__ out, int NG) {
    int i = blockIdx.x * blockDim.x + threadIdx.x;
    if (i < NG) {
        float s = out_b[0];
        #pragma unroll 8
        for (int k = 0; k < 128; ++k) s += g[i * 128 + k] * out_w[k];
        out[i] = s;
    }
}

// ---------------- fused WMMA GEMM ----------------
// Apack: A-fragment-major [tile][KSTEPS][lane][2xuint4].
// W1/W2: B-fragment-major [KSTEPS][4][lane][2xuint4], staged in LDS.
// 128 threads = 4 waves, one 16-row tile per wave.
// MODE 0: Hio = A@W1 + b1
// MODE 1: Hio = softplus(sigmoid(A@W1+b1) * softplus(A@W2+b2) + Hio)
template <int KSTEPS, int NMAT, int MODE>
__global__ void wmma_gemm_kernel(const uint4* __restrict__ Apack,
                                 const uint4* __restrict__ W1,
                                 const uint4* __restrict__ W2,
                                 const float* __restrict__ b1,
                                 const float* __restrict__ b2,
                                 float* __restrict__ Hio, int Nrows) {
    extern __shared__ char smem_raw[];
    uint4* sW = (uint4*)smem_raw;                 // NMAT * KSTEPS * 256 uint4
    constexpr int WQ = KSTEPS * 256;              // uint4 per matrix
    const int tid  = threadIdx.x;
    const int lane = tid & 31;
    const int wv   = tid >> 5;

    __builtin_prefetch(Apack, 0, 3);

    for (int i = tid; i < WQ; i += 128) sW[i] = W1[i];
    if (NMAT == 2)
        for (int i = tid; i < WQ; i += 128) sW[WQ + i] = W2[i];
    __syncthreads();

    const int tile = blockIdx.x * 4 + wv;
    const uint4* ap = Apack + ((size_t)tile * KSTEPS) * 64 + lane * 2;

    v8f acc1[4] = {};
    v8f acc2[4] = {};

    #pragma unroll
    for (int kk = 0; kk < KSTEPS; ++kk) {
        Frag16 a;
        a.q[0] = ap[kk * 64];
        a.q[1] = ap[kk * 64 + 1];
        #pragma unroll
        for (int nb = 0; nb < 4; ++nb) {
            const int base = ((kk * 4 + nb) * 32 + lane) * 2;
            Frag16 b;
            b.q[0] = sW[base];
            b.q[1] = sW[base + 1];
            acc1[nb] = __builtin_amdgcn_wmma_f32_16x16x32_f16(
                false, a.h, false, b.h, (short)0, acc1[nb], false, false);
            if (NMAT == 2) {
                Frag16 c;
                c.q[0] = sW[WQ + base];
                c.q[1] = sW[WQ + base + 1];
                acc2[nb] = __builtin_amdgcn_wmma_f32_16x16x32_f16(
                    false, a.h, false, c.h, (short)0, acc2[nb], false, false);
            }
        }
    }

    // epilogue: C/D layout m = v + 8*(lane>=16), n = nb*16 + lane%16
    const int l16 = lane & 15, h16 = lane >> 4;
    #pragma unroll
    for (int nb = 0; nb < 4; ++nb) {
        #pragma unroll
        for (int v = 0; v < 8; ++v) {
            int row = tile * 16 + v + h16 * 8;
            int col = nb * 16 + l16;
            if (row < Nrows) {
                float* hp = Hio + (size_t)row * HD + col;
                float zf = acc1[nb][v] + b1[col];
                if (MODE == 0) {
                    *hp = zf;
                } else {
                    float zc = acc2[nb][v] + b2[col];
                    float gate = sigmoidf(zf) * softplusf(zc) + *hp;
                    *hp = softplusf(gate);
                }
            }
        }
    }
}

// ---------------- host side ----------------

static inline dim3 nblk(long long n) { return dim3((unsigned)((n + 255) / 256)); }

extern "C" void kernel_launch(void* const* d_in, const int* in_sizes, int n_in,
                              void* d_out, int out_size, void* d_ws, size_t ws_size,
                              hipStream_t stream) {
    const float* x          = (const float*)d_in[0];
    const int*   node_idx   = (const int*)d_in[1];
    const float* hedge_attr = (const float*)d_in[2];
    const int*   batch      = (const int*)d_in[3];
    const float* embed_w    = (const float*)d_in[4];
    const float* embed_b    = (const float*)d_in[5];
    const float* lin_f_w    = (const float*)d_in[6];
    const float* lin_f_b    = (const float*)d_in[7];
    const float* lin_c_w    = (const float*)d_in[8];
    const float* lin_c_b    = (const float*)d_in[9];
    const float* proj_w     = (const float*)d_in[10];
    const float* proj_b     = (const float*)d_in[11];
    const float* out_w      = (const float*)d_in[12];
    const float* out_b      = (const float*)d_in[13];
    float* out = (float*)d_out;

    const int N  = in_sizes[0] / XK;
    const int E  = in_sizes[1] / 2;
    const int H  = in_sizes[2] / HF;
    const int NL = in_sizes[6] / (ZD * HD);
    const int NG = out_size;
    const int* hedge_idx = node_idx + E;

    const long long T    = (N + 15) / 16;        // 16-row tiles
    const long long Tpad = ((T + 3) / 4) * 4;    // multiple of 4 (4 waves/block)

    // workspace bump allocator (256B aligned)
    char* base = (char*)d_ws;
    size_t off = 0;
    auto alloc = [&](size_t bytes) -> void* {
        off = (off + 255) & ~(size_t)255;
        void* p = base + off;
        off += bytes;
        return p;
    };
    float*  h        = (float*)alloc((size_t)N * HD * 4);
    float*  hm       = (float*)alloc((size_t)H * HD * 4);
    float*  za       = (float*)alloc((size_t)N * HF * 4);
    float*  zb       = (float*)alloc((size_t)N * HD * 4);
    unsigned int* zpk  = (unsigned int*)alloc((size_t)Tpad * 1536 * 4); // z A-frag pack
    unsigned int* xpk  = (unsigned int*)alloc((size_t)Tpad * 768 * 4);  // x A-frag pack
    unsigned int* wepk = (unsigned int*)alloc((size_t)3 * 1024 * 4);
    unsigned int* wfpk = (unsigned int*)alloc((size_t)NL * 6 * 1024 * 4);
    unsigned int* wcpk = (unsigned int*)alloc((size_t)NL * 6 * 1024 * 4);
    float*  invHedge = (float*)alloc((size_t)H * 4);
    float*  invNode  = (float*)alloc((size_t)N * 4);
    float*  invBatch = (float*)alloc((size_t)NG * 4);
    float*  pooled   = (float*)alloc((size_t)NG * HD * 4);
    float*  g        = (float*)alloc((size_t)NG * 128 * 4);
    (void)ws_size;

    // ---- degrees ----
    fzero_kernel<<<nblk(H), 256, 0, stream>>>(invHedge, H);
    fzero_kernel<<<nblk(N), 256, 0, stream>>>(invNode, N);
    fzero_kernel<<<nblk(NG), 256, 0, stream>>>(invBatch, NG);
    count_inc_kernel<<<nblk(E), 256, 0, stream>>>(node_idx, hedge_idx, invNode, invHedge, E);
    count_batch_kernel<<<nblk(N), 256, 0, stream>>>(batch, invBatch, N);
    inv_inplace_kernel<<<nblk(H), 256, 0, stream>>>(invHedge, H);
    inv_inplace_kernel<<<nblk(N), 256, 0, stream>>>(invNode, N);
    inv_inplace_kernel<<<nblk(NG), 256, 0, stream>>>(invBatch, NG);

    // ---- fragment packing of constants ----
    cvt_x_pack_kernel<<<nblk(Tpad * 768), 256, 0, stream>>>(x, xpk, N, Tpad * 768);
    cvt_w_pack_kernel<<<nblk(3 * 1024), 256, 0, stream>>>(embed_w, wepk, XK, 3);
    for (int l = 0; l < NL; ++l) {
        cvt_w_pack_kernel<<<nblk(6 * 1024), 256, 0, stream>>>(
            lin_f_w + (size_t)l * ZD * HD, wfpk + (size_t)l * 6144, ZD, 6);
        cvt_w_pack_kernel<<<nblk(6 * 1024), 256, 0, stream>>>(
            lin_c_w + (size_t)l * ZD * HD, wcpk + (size_t)l * 6144, ZD, 6);
    }

    // ---- embed: h = x @ We + b ----
    {
        size_t shmem = (size_t)1 * 3 * 256 * 16;   // 12 KB
        wmma_gemm_kernel<3, 1, 0><<<dim3((unsigned)(Tpad / 4)), 128, shmem, stream>>>(
            (const uint4*)xpk, (const uint4*)wepk, (const uint4*)nullptr,
            embed_b, (const float*)nullptr, h, N);
    }

    // ---- NL conv layers ----
    for (int l = 0; l < NL; ++l) {
        fzero_kernel<<<nblk((long long)H * HD), 256, 0, stream>>>(hm, (long long)H * HD);
        scatter_hm_kernel<<<nblk((long long)E * HD), 256, 0, stream>>>(
            node_idx, hedge_idx, h, hm, (long long)E * HD);
        rowscale_kernel<<<nblk((long long)H * HD), 256, 0, stream>>>(
            hm, invHedge, (long long)H * HD, HD);

        fzero_kernel<<<nblk((long long)N * HF), 256, 0, stream>>>(za, (long long)N * HF);
        fzero_kernel<<<nblk((long long)N * HD), 256, 0, stream>>>(zb, (long long)N * HD);
        scatter_z_kernel<<<nblk((long long)E * 128), 256, 0, stream>>>(
            node_idx, hedge_idx, hedge_attr, hm, za, zb, (long long)E * 128);

        build_z_pack_kernel<<<nblk(Tpad * 1536), 256, 0, stream>>>(
            h, za, zb, invNode, zpk, N, Tpad * 1536);

        size_t shmem = (size_t)2 * 6 * 256 * 16;   // 48 KB
        wmma_gemm_kernel<6, 2, 1><<<dim3((unsigned)(Tpad / 4)), 128, shmem, stream>>>(
            (const uint4*)zpk, (const uint4*)(wfpk + (size_t)l * 6144),
            (const uint4*)(wcpk + (size_t)l * 6144),
            lin_f_b + (size_t)l * HD, lin_c_b + (size_t)l * HD, h, N);
    }

    // ---- readout ----
    fzero_kernel<<<nblk((long long)NG * HD), 256, 0, stream>>>(pooled, (long long)NG * HD);
    pool_scatter_kernel<<<nblk((long long)N * HD), 256, 0, stream>>>(
        batch, h, pooled, (long long)N * HD);
    g_kernel<<<nblk(NG * 128), 256, 0, stream>>>(pooled, invBatch, proj_w, proj_b, g, NG * 128);
    out_kernel<<<nblk(NG), 256, 0, stream>>>(g, out_w, out_b, out, NG);
}
